// DeepConvGraphEncoderDownstream_36618891166132
// MI455X (gfx1250) — compile-verified, matrix operand order checked by reading, not run on hardware
//
#include <hip/hip_runtime.h>
#include <hip/hip_bf16.h>
#include <math.h>

// ---------------------------------------------------------------------------
// Problem constants (from reference)
// ---------------------------------------------------------------------------
#define BB   64
#define TT   256
#define NN   24          // graph nodes
#define FF   6           // input features
#define HH   256         // lstm hidden
#define EMB  128
#define WIN  32
#define STR  2
#define NWIN 113         // (256-32)/2 + 1
#define NEDGE 96
#define ETOT 120         // + self loops
#define ROWS (BB*TT*NN)  // 393216 frame-node rows
#define FRAMES (BB*TT)   // 16384

typedef __attribute__((ext_vector_type(16))) _Float16 v16h;
typedef __attribute__((ext_vector_type(8)))  _Float16 v8h;
typedef __attribute__((ext_vector_type(8)))  float    v8f;
typedef __attribute__((ext_vector_type(4)))  float    v4f;

union HV { v16h v16; v8h v8[2]; };

// Load a 16x32 f16 fragment (A or B^T layout): lane l covers row (l&15),
// k-chunks at k0 + ((l>>4)<<3) and +16 (two 16-byte loads).
__device__ __forceinline__ v16h load_frag(const _Float16* base, int row, int ld,
                                          int k0, int lane) {
  const _Float16* p = base + (size_t)row * ld + k0 + ((lane >> 4) << 3);
  HV u;
  u.v8[0] = *(const v8h*)p;
  u.v8[1] = *(const v8h*)(p + 16);
  return u.v16;
}

__device__ __forceinline__ v8f wmma32(v16h a, v16h b, v8f c) {
  return __builtin_amdgcn_wmma_f32_16x16x32_f16(false, a, false, b,
                                                (short)0, c, false, false);
}

__device__ __forceinline__ float sigf(float x) {
  return 1.0f / (1.0f + __expf(-x));
}

// branchless tanh: 1 - 2/(exp(2x)+1); saturates to +/-1 as exp -> inf/0
__device__ __forceinline__ float tanh_fast(float x) {
  return 1.0f - 2.0f / (__expf(2.0f * x) + 1.0f);
}

// ---------------------------------------------------------------------------
// Prep kernels
// ---------------------------------------------------------------------------
__global__ void gcn_norm_kernel(const int* __restrict__ ei,
                                int* __restrict__ srcN, int* __restrict__ dstN,
                                float* __restrict__ normv) {
  __shared__ float deg[NN];
  int t = threadIdx.x;
  if (t < NN) deg[t] = 0.0f;
  __syncthreads();
  if (t < ETOT) {
    int s = (t < NEDGE) ? ei[t] : (t - NEDGE);
    int d = (t < NEDGE) ? ei[NEDGE + t] : (t - NEDGE);
    srcN[t] = s; dstN[t] = d;
    atomicAdd(&deg[d], 1.0f);
  }
  __syncthreads();
  if (t < ETOT) {
    float dd = deg[dstN[t]], ds = deg[srcN[t]];
    float id = dd > 0.0f ? rsqrtf(dd) : 0.0f;
    float is = ds > 0.0f ? rsqrtf(ds) : 0.0f;
    normv[t] = id * is;
  }
}

__global__ void cvt_f16_kernel(const float* __restrict__ in,
                               _Float16* __restrict__ out, int n) {
  int i = blockIdx.x * blockDim.x + threadIdx.x;
  if (i < n) out[i] = (_Float16)in[i];
}

// out[n*Kpad + k] = (k<K) ? W[k*Nw + n] : 0   (transpose + K-pad + f16)
__global__ void transpose_pad_kernel(const float* __restrict__ w,
                                     _Float16* __restrict__ out,
                                     int K, int Nw, int Kpad) {
  int i = blockIdx.x * blockDim.x + threadIdx.x;
  if (i >= Nw * Kpad) return;
  int n = i / Kpad, k = i % Kpad;
  out[i] = (k < K) ? (_Float16)w[k * Nw + n] : (_Float16)0.0f;
}

__global__ void add_bias_kernel(const float* __restrict__ a,
                                const float* __restrict__ b,
                                float* __restrict__ out, int n) {
  int i = blockIdx.x * blockDim.x + threadIdx.x;
  if (i < n) out[i] = a[i] + b[i];
}

// data [ROWS, 6] f32 -> X0 [ROWS, 32] f16 zero-padded
__global__ void pad_input_kernel(const float* __restrict__ data,
                                 _Float16* __restrict__ X0) {
  int i = blockIdx.x * blockDim.x + threadIdx.x;
  if (i >= ROWS * 32) return;
  int row = i >> 5, col = i & 31;
  X0[i] = (col < FF) ? (_Float16)data[(size_t)row * FF + col] : (_Float16)0.0f;
}

// ---------------------------------------------------------------------------
// Generic WMMA GEMM: C[M,N] = A[M,K]f16 * Bt[N,K]f16 (+bias), f32 out.
// grid.x = M/16, grid.y = ceil((N/16)/8); 8 waves/block, one n-tile per wave.
// GATE_PACK (for LSTM pre-activations, N=1024): store row-major with the 4
// gate values of a hidden unit contiguous: C[m][ (n&255)*4 + (n>>8) ].
// ---------------------------------------------------------------------------
template <bool BIAS, bool GATE_PACK>
__global__ void gemm_f16_kernel(const _Float16* __restrict__ A,
                                const _Float16* __restrict__ Bt,
                                float* __restrict__ C,
                                const float* __restrict__ bias,
                                int N, int K) {
  int lane = threadIdx.x & 31;
  int wave = threadIdx.x >> 5;
  int mt = blockIdx.x;
  int nt = blockIdx.y * 8 + wave;
  if (nt * 16 >= N) return;

  int arow = mt * 16 + (lane & 15);
  int brow = nt * 16 + (lane & 15);
  v8f acc = {};
  for (int k0 = 0; k0 < K; k0 += 32) {
    __builtin_prefetch(A + (size_t)arow * K + k0 + 64, 0, 1);
    v16h a = load_frag(A,  arow, K, k0, lane);
    v16h b = load_frag(Bt, brow, K, k0, lane);
    acc = wmma32(a, b, acc);
  }
  int n = nt * 16 + (lane & 15);
  int mbase = mt * 16 + ((lane >> 4) << 3);
  float bv = BIAS ? bias[n] : 0.0f;
  int ncol = GATE_PACK ? ((n & 255) * 4 + (n >> 8)) : n;
#pragma unroll
  for (int r = 0; r < 8; ++r)
    C[(size_t)(mbase + r) * N + ncol] = acc[r] + bv;
}

// ---------------------------------------------------------------------------
// GCN aggregation: out[f,d,:] = relu( sum_e norm_e * G[f,src_e,:] + bias )
// one block per frame; blockDim = Fw; serial edge loop per feature -> no races
// ---------------------------------------------------------------------------
__global__ void gcn_aggregate_kernel(const float* __restrict__ G,
                                     const int* __restrict__ srcN,
                                     const int* __restrict__ dstN,
                                     const float* __restrict__ normv,
                                     const float* __restrict__ bias,
                                     _Float16* __restrict__ out, int Fw) {
  extern __shared__ float acc[];            // [NN * Fw]
  __shared__ int   sS[ETOT], sD[ETOT];
  __shared__ float sN[ETOT];
  int tid = threadIdx.x;
  for (int e = tid; e < ETOT; e += blockDim.x) {
    sS[e] = srcN[e]; sD[e] = dstN[e]; sN[e] = normv[e];
  }
#pragma unroll 4
  for (int n = 0; n < NN; ++n) acc[n * Fw + tid] = 0.0f;
  __syncthreads();

  const float* g = G + (size_t)blockIdx.x * NN * Fw;
  for (int e = 0; e < ETOT; ++e)
    acc[sD[e] * Fw + tid] += sN[e] * g[sS[e] * Fw + tid];

  float bv = bias[tid];
  _Float16* o = out + (size_t)blockIdx.x * NN * Fw;
#pragma unroll 4
  for (int n = 0; n < NN; ++n) {
    float v = acc[n * Fw + tid] + bv;
    o[n * Fw + tid] = (_Float16)(v > 0.0f ? v : 0.0f);
  }
}

// mean over nodes: x_all[f, j] = mean_n H4[f*24+n, j]
__global__ void node_mean_kernel(const _Float16* __restrict__ H4,
                                 _Float16* __restrict__ x_all) {
  int f = blockIdx.x, j = threadIdx.x;     // blockDim = 256
  const _Float16* p = H4 + (size_t)f * NN * HH + j;
  float s = 0.0f;
#pragma unroll
  for (int n = 0; n < NN; ++n) s += (float)p[n * HH];
  x_all[(size_t)f * HH + j] = (_Float16)(s * (1.0f / (float)NN));
}

// ---------------------------------------------------------------------------
// Fused LSTM recurrence: one block per (window, direction), 8 waves.
// h double-buffered in LDS (f16); c in VGPRs (fully-unrolled q => constant
// indices, no private-array scratch); gates via WMMA on Whh^T.
// Xg = x@Wih^T + (bih+bhh) precomputed for all (b,t), gate-packed:
//   Xg[(b*T+t)*1024 + j*4 + g]  with g in {i,f,g,o}.
// Whh fragments are deliberately re-streamed from L2 each step; an opaque
// zero offset blocks LICM from hoisting (and spilling) them across steps,
// and a scheduling fence between q-bodies bounds in-flight registers.
// ---------------------------------------------------------------------------
__global__ void __launch_bounds__(256, 1)
lstm_kernel(const _Float16* __restrict__ WhhF,
            const _Float16* __restrict__ WhhB,
            const float* __restrict__ XgF,
            const float* __restrict__ XgB,
            _Float16* __restrict__ Lbuf) {
  __shared__ _Float16 hbuf[2][BB * HH];     // 2 x 32 KB
  int win = blockIdx.x, dir = blockIdx.y;
  int start = win * STR;
  const _Float16* Whh = dir ? WhhB : WhhF;  // [4H=1024, H=256] f16 (= B^T)
  const float*    Xg  = dir ? XgB  : XgF;   // [B*T, 256, 4] f32 (bias folded)

  int lane = threadIdx.x & 31;
  int wave = threadIdx.x >> 5;

  for (int i = threadIdx.x; i < BB * HH; i += blockDim.x)
    hbuf[0][i] = (_Float16)0.0f;
  float creg[8][8];
#pragma unroll
  for (int q = 0; q < 8; ++q)
#pragma unroll
    for (int r = 0; r < 8; ++r) creg[q][r] = 0.0f;
  __syncthreads();

#pragma unroll 1
  for (int t = 0; t < WIN; ++t) {
    int cur = t & 1, nxt = cur ^ 1;
    int tt = dir ? (start + WIN - 1 - t) : (start + t);
    // Opaque zero: compiler must assume Wt differs per step, so the 256
    // loop-invariant B fragments are NOT hoisted/spilled across steps.
    size_t zoff = 0;
    asm volatile("" : "+s"(zoff));
    const _Float16* Wt = Whh + zoff;
#pragma unroll
    for (int q = 0; q < 8; ++q) {
      int bid = wave * 8 + q;               // 64 blocks of 16x16
      int mt = bid >> 4;                    // batch tile  (4)
      int jt = bid & 15;                    // hidden tile (16)
      v8f aI = {}, aF = {}, aG = {}, aO = {};
      int arow = mt * 16 + (lane & 15);
      int brow = jt * 16 + (lane & 15);
#pragma unroll 2
      for (int k0 = 0; k0 < HH; k0 += 32) {
        v16h a  = load_frag(&hbuf[cur][0], arow, HH, k0, lane);
        v16h bi = load_frag(Wt, 0 * HH + brow, HH, k0, lane);
        v16h bf = load_frag(Wt, 1 * HH + brow, HH, k0, lane);
        v16h bg = load_frag(Wt, 2 * HH + brow, HH, k0, lane);
        v16h bo = load_frag(Wt, 3 * HH + brow, HH, k0, lane);
        aI = wmma32(a, bi, aI);
        aF = wmma32(a, bf, aF);
        aG = wmma32(a, bg, aG);
        aO = wmma32(a, bo, aO);
      }
      int j = jt * 16 + (lane & 15);
      int mbase = mt * 16 + ((lane >> 4) << 3);
#pragma unroll
      for (int r = 0; r < 8; ++r) {
        int b = mbase + r;
        // one aligned 16B load: {i, f, g, o} pre-activations for (b, tt, j)
        v4f xv = *(const v4f*)(Xg + ((size_t)b * TT + tt) * (4 * HH) + j * 4);
        float gi = aI[r] + xv.x;
        float gf = aF[r] + xv.y;
        float gg = aG[r] + xv.z;
        float go = aO[r] + xv.w;
        float c  = sigf(gf) * creg[q][r] + sigf(gi) * tanh_fast(gg);
        creg[q][r] = c;
        hbuf[nxt][b * HH + j] = (_Float16)(sigf(go) * tanh_fast(c));
      }
      // Scheduling fence: keep each q-body's fragment loads local so the
      // unrolled step body doesn't balloon register pressure.
      asm volatile("" ::: "memory");
    }
    __syncthreads();
  }
  // final h lives in hbuf[0] after 32 steps (t=31 wrote buffer 0)
  for (int i = threadIdx.x; i < BB * HH; i += blockDim.x) {
    int b = i >> 8, j = i & 255;
    Lbuf[((size_t)(win * BB + b)) * (2 * HH) + dir * HH + j] = hbuf[0][i];
  }
}

// ---------------------------------------------------------------------------
// FC: out[b, win, :] = L[win*64+b, :512] @ WfcT^T + bfc   (M=7232,K=512,N=128)
// ---------------------------------------------------------------------------
__global__ void fc_kernel(const _Float16* __restrict__ L,
                          const _Float16* __restrict__ WfcT,  // [128, 512]
                          const float* __restrict__ bfc,
                          float* __restrict__ out) {
  int lane = threadIdx.x & 31;
  int wave = threadIdx.x >> 5;              // 8 waves = 8 n-tiles (N=128)
  int mt = blockIdx.x;
  int arow = mt * 16 + (lane & 15);
  int brow = wave * 16 + (lane & 15);
  v8f acc = {};
  for (int k0 = 0; k0 < 2 * HH; k0 += 32) {
    v16h a = load_frag(L,    arow, 2 * HH, k0, lane);
    v16h b = load_frag(WfcT, brow, 2 * HH, k0, lane);
    acc = wmma32(a, b, acc);
  }
  int n = wave * 16 + (lane & 15);
  int mbase = mt * 16 + ((lane >> 4) << 3);
  float bv = bfc[n];
#pragma unroll
  for (int r = 0; r < 8; ++r) {
    int m = mbase + r;
    int w = m >> 6, b = m & 63;             // row = win*64 + b
    out[((size_t)b * NWIN + w) * EMB + n] = acc[r] + bv;
  }
}

// ---------------------------------------------------------------------------
// Launcher
// ---------------------------------------------------------------------------
extern "C" void kernel_launch(void* const* d_in, const int* in_sizes, int n_in,
                              void* d_out, int out_size, void* d_ws, size_t ws_size,
                              hipStream_t stream) {
  (void)in_sizes; (void)n_in; (void)out_size; (void)ws_size;
  const float* data = (const float*)d_in[0];
  const int*   ei   = (const int*)  d_in[1];
  const float* W1 = (const float*)d_in[2],  *b1 = (const float*)d_in[3];
  const float* W2 = (const float*)d_in[4],  *b2 = (const float*)d_in[5];
  const float* W3 = (const float*)d_in[6],  *b3 = (const float*)d_in[7];
  const float* W4 = (const float*)d_in[8],  *b4 = (const float*)d_in[9];
  const float* WihF = (const float*)d_in[10], *WhhF = (const float*)d_in[11];
  const float* bihF = (const float*)d_in[12], *bhhF = (const float*)d_in[13];
  const float* WihB = (const float*)d_in[14], *WhhB = (const float*)d_in[15];
  const float* bihB = (const float*)d_in[16], *bhhB = (const float*)d_in[17];
  const float* Wfc  = (const float*)d_in[18], *bfc  = (const float*)d_in[19];
  float* out = (float*)d_out;

  size_t off = 0;
  auto alloc = [&](size_t bytes) -> void* {
    off = (off + 255) & ~(size_t)255;
    void* p = (char*)d_ws + off;
    off += bytes;
    return p;
  };
  int*      srcN  = (int*)alloc(ETOT * 4);
  int*      dstN  = (int*)alloc(ETOT * 4);
  float*    normv = (float*)alloc(ETOT * 4);
  _Float16* W1t   = (_Float16*)alloc((size_t)64  * 32  * 2);
  _Float16* W2t   = (_Float16*)alloc((size_t)128 * 64  * 2);
  _Float16* W3t   = (_Float16*)alloc((size_t)256 * 128 * 2);
  _Float16* W4t   = (_Float16*)alloc((size_t)256 * 256 * 2);
  _Float16* WihF6 = (_Float16*)alloc((size_t)1024 * 256 * 2);
  _Float16* WhhF6 = (_Float16*)alloc((size_t)1024 * 256 * 2);
  _Float16* WihB6 = (_Float16*)alloc((size_t)1024 * 256 * 2);
  _Float16* WhhB6 = (_Float16*)alloc((size_t)1024 * 256 * 2);
  _Float16* WfcT  = (_Float16*)alloc((size_t)128 * 512 * 2);
  float*    biasF = (float*)alloc(1024 * 4);
  float*    biasB = (float*)alloc(1024 * 4);
  _Float16* X0    = (_Float16*)alloc((size_t)ROWS * 32 * 2);
  _Float16* bufA  = (_Float16*)alloc((size_t)ROWS * 256 * 2);
  _Float16* bufB  = (_Float16*)alloc((size_t)ROWS * 256 * 2);
  float*    bufG  = (float*)alloc((size_t)ROWS * 256 * 4);
  _Float16* xall  = (_Float16*)alloc((size_t)FRAMES * 256 * 2);
  float*    XgF   = (float*)alloc((size_t)FRAMES * 1024 * 4);
  float*    XgB   = (float*)alloc((size_t)FRAMES * 1024 * 4);
  _Float16* Lbuf  = (_Float16*)alloc((size_t)NWIN * BB * 512 * 2);

  // --- prep ---
  gcn_norm_kernel<<<1, 128, 0, stream>>>(ei, srcN, dstN, normv);
  pad_input_kernel<<<(ROWS * 32 + 255) / 256, 256, 0, stream>>>(data, X0);
  transpose_pad_kernel<<<(64 * 32 + 255) / 256, 256, 0, stream>>>(W1, W1t, 6, 64, 32);
  transpose_pad_kernel<<<(128 * 64 + 255) / 256, 256, 0, stream>>>(W2, W2t, 64, 128, 64);
  transpose_pad_kernel<<<(256 * 128 + 255) / 256, 256, 0, stream>>>(W3, W3t, 128, 256, 128);
  transpose_pad_kernel<<<(256 * 256 + 255) / 256, 256, 0, stream>>>(W4, W4t, 256, 256, 256);
  transpose_pad_kernel<<<(128 * 512 + 255) / 256, 256, 0, stream>>>(Wfc, WfcT, 512, 128, 512);
  cvt_f16_kernel<<<(262144 + 255) / 256, 256, 0, stream>>>(WihF, WihF6, 262144);
  cvt_f16_kernel<<<(262144 + 255) / 256, 256, 0, stream>>>(WhhF, WhhF6, 262144);
  cvt_f16_kernel<<<(262144 + 255) / 256, 256, 0, stream>>>(WihB, WihB6, 262144);
  cvt_f16_kernel<<<(262144 + 255) / 256, 256, 0, stream>>>(WhhB, WhhB6, 262144);
  add_bias_kernel<<<4, 256, 0, stream>>>(bihF, bhhF, biasF, 1024);
  add_bias_kernel<<<4, 256, 0, stream>>>(bihB, bhhB, biasB, 1024);

  const int MT = ROWS / 16;  // 24576 m-tiles for GCN GEMMs
  // --- GCN layer 1: [ROWS,32] x [32,64] ---
  gemm_f16_kernel<false, false><<<dim3(MT, 1), 256, 0, stream>>>(X0, W1t, bufG, nullptr, 64, 32);
  gcn_aggregate_kernel<<<FRAMES, 64, NN * 64 * 4, stream>>>(bufG, srcN, dstN, normv, b1, bufA, 64);
  // --- layer 2: [ROWS,64] x [64,128] ---
  gemm_f16_kernel<false, false><<<dim3(MT, 1), 256, 0, stream>>>(bufA, W2t, bufG, nullptr, 128, 64);
  gcn_aggregate_kernel<<<FRAMES, 128, NN * 128 * 4, stream>>>(bufG, srcN, dstN, normv, b2, bufB, 128);
  // --- layer 3: [ROWS,128] x [128,256] ---
  gemm_f16_kernel<false, false><<<dim3(MT, 2), 256, 0, stream>>>(bufB, W3t, bufG, nullptr, 256, 128);
  gcn_aggregate_kernel<<<FRAMES, 256, NN * 256 * 4, stream>>>(bufG, srcN, dstN, normv, b3, bufA, 256);
  // --- layer 4: [ROWS,256] x [256,256] ---
  gemm_f16_kernel<false, false><<<dim3(MT, 2), 256, 0, stream>>>(bufA, W4t, bufG, nullptr, 256, 256);
  gcn_aggregate_kernel<<<FRAMES, 256, NN * 256 * 4, stream>>>(bufG, srcN, dstN, normv, b4, bufB, 256);
  // --- node mean -> x_all [FRAMES, 256] f16 ---
  node_mean_kernel<<<FRAMES, 256, 0, stream>>>(bufB, xall);

  // --- LSTM input gates for all timesteps: Xg = x@Wih^T + (bih+bhh),
  //     gate-packed [frame][j][4] ---
  gemm_f16_kernel<true, true><<<dim3(FRAMES / 16, 8), 256, 0, stream>>>(xall, WihF6, XgF, biasF, 1024, 256);
  gemm_f16_kernel<true, true><<<dim3(FRAMES / 16, 8), 256, 0, stream>>>(xall, WihB6, XgB, biasB, 1024, 256);

  // --- recurrence over 113 windows x 2 directions ---
  lstm_kernel<<<dim3(NWIN, 2), 256, 0, stream>>>(WhhF6, WhhB6, XgF, XgB, Lbuf);

  // --- final FC with [B, win, EMB] scatter ---
  fc_kernel<<<(NWIN * BB) / 16, 256, 0, stream>>>(Lbuf, WfcT, bfc, out);
}
